// FactorModel_49804440765051
// MI455X (gfx1250) — matrix operand order repaired and checked by reading, not compiled, not run on hardware
//
#include <hip/hip_runtime.h>

// ---------------------------------------------------------------------------
// Problem constants (from the reference)
// ---------------------------------------------------------------------------
#define BATCH   16384
#define DIN     1024
#define DH      2048
#define NN      4
#define LOWERC  0.05f
#define UPPERC  0.5f
#define C_VAL   0.1f
#define NEWTON  25

// GEMM tiling
#define MTILE   32          // rows of x per block
#define SM      2           // 16-row sub-tiles per block (MTILE/16)
#define NWAVES  8           // 256 threads / wave32
#define TBLK    4           // h-tiles processed concurrently per wave
#define PITCH   1032        // LDS row pitch in bf16 elems (1024 + 8 pad -> bank-conflict free)

// ---------------------------------------------------------------------------
// Types for CDNA5 WMMA
// ---------------------------------------------------------------------------
typedef __bf16 bf16_t;
typedef bf16_t         v16bf __attribute__((ext_vector_type(16)));
typedef float          v8f   __attribute__((ext_vector_type(8)));
typedef unsigned short v16us __attribute__((ext_vector_type(16)));

union FragU { v16us u; uint4 q[2]; };

__device__ __forceinline__ unsigned short f2bfbits(float f) {
    unsigned u = __builtin_bit_cast(unsigned, f);
    u += 0x7FFFu + ((u >> 16) & 1u);           // round-to-nearest-even
    return (unsigned short)(u >> 16);
}
__device__ __forceinline__ float bfbits2f(unsigned short h) {
    unsigned u = ((unsigned)h) << 16;
    return __builtin_bit_cast(float, u);
}

// ---------------------------------------------------------------------------
// Kernel 0: pack W1 (fp32, DIN x DH row-major) -> bf16 hi/lo, transposed and
// blocked as [c/16][k/16][c%16][k%16] so a WMMA B-fragment is one contiguous
// 32B read per lane and 512B contiguous per 16-lane group.
// ---------------------------------------------------------------------------
__global__ void pack_w1_kernel(const float* __restrict__ W1,
                               unsigned short* __restrict__ hiP,
                               unsigned short* __restrict__ loP) {
    int idx = blockIdx.x * 256 + threadIdx.x;
    if (idx >= DIN * DH) return;
    int k = idx >> 11;            // DH = 2048
    int c = idx & (DH - 1);
    float f = W1[idx];
    unsigned short hb = f2bfbits(f);
    unsigned short lb = f2bfbits(f - bfbits2f(hb));
    size_t dst = ((size_t)(c >> 4) * (DIN / 16) + (size_t)(k >> 4)) * 256
               + (size_t)(c & 15) * 16 + (size_t)(k & 15);
    hiP[dst] = hb;
    loP[dst] = lb;
}

// ---------------------------------------------------------------------------
// Kernel 1: fused  logits = LeakyReLU(x@W1 + b1) @ W2 + b2
//           -> softmax -> clip -> row min-max  ==> b  (BATCH x 4)
// bf16 WMMA with 3-term hi/lo split for ~fp32 accuracy.
// TBLK h-tiles share one set of A fragments -> 24 WMMA : 8 ds : 16 vmem per
// k-chunk, enough independent accumulators to hide the WMMA WAR hazard.
// ---------------------------------------------------------------------------
__global__ void __launch_bounds__(256)
fused_logits_kernel(const float* __restrict__ x,
                    const unsigned short* __restrict__ w1hi,
                    const unsigned short* __restrict__ w1lo,
                    const float* __restrict__ b1,
                    const float* __restrict__ W2,
                    const float* __restrict__ b2,
                    float* __restrict__ bvec) {
    extern __shared__ char smem[];
    unsigned short* sHi = (unsigned short*)smem;            // MTILE*PITCH bf16
    unsigned short* sLo = sHi + MTILE * PITCH;              // MTILE*PITCH bf16
    float* sLog = (float*)(sLo + MTILE * PITCH);            // MTILE*NN f32

    const int tid  = threadIdx.x;
    const int row0 = blockIdx.x * MTILE;

    // ---- stage x tile into LDS as bf16 hi/lo (coalesced fp32 reads) ----
    for (int i = tid; i < MTILE * DIN; i += 256) {
        int r = i >> 10;                 // DIN = 1024
        int k = i & (DIN - 1);
        float f = x[(size_t)(row0 + r) * DIN + k];
        unsigned short hb = f2bfbits(f);
        sHi[r * PITCH + k] = hb;
        sLo[r * PITCH + k] = f2bfbits(f - bfbits2f(hb));
    }
    if (tid < MTILE * NN) sLog[tid] = 0.0f;
    __syncthreads();

    const int lane = tid & 31;
    const int wave = tid >> 5;
    const int rA   = lane & 15;                 // A-matrix row for this lane
    const int kbA  = (lane < 16) ? 0 : 16;      // A K-offset (contiguous k = 16h + s, matches B)
    const int kbB  = (lane < 16) ? 0 : 16;      // B K-offset (contiguous k = 16h + s, matches A)
    const int ci   = lane & 15;                 // B column within tile

    // per-lane row bases: keeps every ds_load offset within the 16-bit imm
    const unsigned short* rowHi = sHi + rA * PITCH;
    const unsigned short* rowLo = sLo + rA * PITCH;

    float lac[SM][8][NN];                       // logit partial sums
    #pragma unroll
    for (int m = 0; m < SM; ++m)
        #pragma unroll
        for (int r = 0; r < 8; ++r)
            #pragma unroll
            for (int n = 0; n < NN; ++n) lac[m][r][n] = 0.0f;

    // wave handles tiles t = wave*16 + g*TBLK + j   (t in [16*wave, 16*wave+15])
    #pragma unroll 1
    for (int g = 0; g < (DH / 16) / (NWAVES * TBLK); ++g) {
        const int tbase = wave * 16 + g * TBLK;

        v8f acc[TBLK][SM];
        #pragma unroll
        for (int j = 0; j < TBLK; ++j)
            #pragma unroll
            for (int m = 0; m < SM; ++m)
                #pragma unroll
                for (int r = 0; r < 8; ++r) acc[j][m][r] = 0.0f;

        #pragma unroll 1
        for (int k0 = 0; k0 < DIN; k0 += 32) {
            const int kB = (k0 + kbB) >> 4;

            // ---- A fragments (shared across the TBLK h-tiles) ----
            v16bf Ah[SM], Al[SM];
            #pragma unroll
            for (int m = 0; m < SM; ++m) {
                const unsigned short* ar = rowHi + m * 16 * PITCH + k0 + kbA;
                const unsigned short* al = rowLo + m * 16 * PITCH + k0 + kbA;
                FragU fh, fl;
                fh.q[0] = *(const uint4*)(ar);
                fh.q[1] = *(const uint4*)(ar + 8);
                fl.q[0] = *(const uint4*)(al);
                fl.q[1] = *(const uint4*)(al + 8);
                Ah[m] = __builtin_bit_cast(v16bf, fh.u);
                Al[m] = __builtin_bit_cast(v16bf, fl.u);
            }

            // ---- B fragments per h-tile, 6 WMMAs each ----
            #pragma unroll
            for (int j = 0; j < TBLK; ++j) {
                const size_t cb = ((size_t)(tbase + j) * (DIN / 16) + kB) * 256
                                + (size_t)ci * 16;
                const uint4* bph = (const uint4*)(w1hi + cb);
                const uint4* bpl = (const uint4*)(w1lo + cb);
                if (j == 0) __builtin_prefetch(bph + 32, 0, 3);   // next K blocks
                FragU bh, bl;
                bh.q[0] = bph[0]; bh.q[1] = bph[1];
                bl.q[0] = bpl[0]; bl.q[1] = bpl[1];
                v16bf Bh = __builtin_bit_cast(v16bf, bh.u);
                v16bf Bl = __builtin_bit_cast(v16bf, bl.u);

                #pragma unroll
                for (int m = 0; m < SM; ++m) {
                    // 3-term split: hi*hi + lo*hi + hi*lo (~fp32 accuracy)
                    acc[j][m] = __builtin_amdgcn_wmma_f32_16x16x32_bf16(false, Ah[m], false, Bh, (short)0, acc[j][m], false, false);
                    acc[j][m] = __builtin_amdgcn_wmma_f32_16x16x32_bf16(false, Al[m], false, Bh, (short)0, acc[j][m], false, false);
                    acc[j][m] = __builtin_amdgcn_wmma_f32_16x16x32_bf16(false, Ah[m], false, Bl, (short)0, acc[j][m], false, false);
                }
            }
        }

        // epilogue: + b1, LeakyReLU(slope -0.1), fold into logits via W2
        #pragma unroll
        for (int j = 0; j < TBLK; ++j) {
            const int jcol = (tbase + j) * 16 + ci;
            const float bias = b1[jcol];
            const float4 w2v = *(const float4*)(W2 + (size_t)jcol * NN);
            #pragma unroll
            for (int m = 0; m < SM; ++m) {
                #pragma unroll
                for (int r = 0; r < 8; ++r) {
                    float h = acc[j][m][r] + bias;
                    h = (h > 0.0f) ? h : (-0.1f * h);
                    lac[m][r][0] += h * w2v.x;
                    lac[m][r][1] += h * w2v.y;
                    lac[m][r][2] += h * w2v.z;
                    lac[m][r][3] += h * w2v.w;
                }
            }
        }
    }

    // ---- reduce logits: butterfly over 16 lanes, ds_add across waves ----
    #pragma unroll
    for (int m = 0; m < SM; ++m)
        #pragma unroll
        for (int r = 0; r < 8; ++r)
            #pragma unroll
            for (int n = 0; n < NN; ++n) {
                float v = lac[m][r][n];
                v += __shfl_xor(v, 1, 32);
                v += __shfl_xor(v, 2, 32);
                v += __shfl_xor(v, 4, 32);
                v += __shfl_xor(v, 8, 32);
                lac[m][r][n] = v;
            }
    if ((lane & 15) == 0) {
        const int rb = (lane < 16) ? 0 : 8;   // C-layout: lanes 0-15 rows r, 16-31 rows r+8
        #pragma unroll
        for (int m = 0; m < SM; ++m)
            #pragma unroll
            for (int r = 0; r < 8; ++r)
                #pragma unroll
                for (int n = 0; n < NN; ++n)
                    atomicAdd(&sLog[(m * 16 + rb + r) * NN + n], lac[m][r][n]);
    }
    __syncthreads();

    // ---- softmax -> clip -> row min-max, one thread per row ----
    if (tid < MTILE) {
        float l[NN], p[NN];
        float mx = -3.0e38f;
        #pragma unroll
        for (int n = 0; n < NN; ++n) { l[n] = sLog[tid * NN + n] + b2[n]; mx = fmaxf(mx, l[n]); }
        float s = 0.0f;
        #pragma unroll
        for (int n = 0; n < NN; ++n) { p[n] = expf(l[n] - mx); s += p[n]; }
        float inv = 1.0f / s;
        float pmin = 3.0e38f, pmax = -3.0e38f;
        #pragma unroll
        for (int n = 0; n < NN; ++n) {
            float pv = p[n] * inv;
            pv = fminf(fmaxf(pv, LOWERC), UPPERC);          // Hardtanh / clip
            p[n] = pv;
            pmin = fminf(pmin, pv);
            pmax = fmaxf(pmax, pv);
        }
        float innv = 1.0f / (pmax - pmin + 1e-8f);
        #pragma unroll
        for (int n = 0; n < NN; ++n)
            bvec[(size_t)(row0 + tid) * NN + n] = (p[n] - pmin) * innv;
    }
}

// ---------------------------------------------------------------------------
// Kernel 2: per-row KKT Newton solve (25 iters, pivoted 5x5), then normalize.
// One thread per batch row; all indexing static -> registers only.
// ---------------------------------------------------------------------------
__global__ void solve_kernel(const float* __restrict__ Qs,
                             const float* __restrict__ bvec,
                             float* __restrict__ out) {
    int row = blockIdx.x * blockDim.x + threadIdx.x;
    if (row >= BATCH) return;

    float qs[4][4];
    #pragma unroll
    for (int k = 0; k < 4; ++k)
        #pragma unroll
        for (int i = 0; i < 4; ++i)
            qs[k][i] = Qs[(size_t)row * 16 + k * 4 + i];

    float Q[4][4];                                          // Q = Qs^T Qs
    #pragma unroll
    for (int i = 0; i < 4; ++i)
        #pragma unroll
        for (int j = 0; j < 4; ++j) {
            float s = 0.0f;
            #pragma unroll
            for (int k = 0; k < 4; ++k) s += qs[k][i] * qs[k][j];
            Q[i][j] = s;
        }

    float bb[4];
    #pragma unroll
    for (int n = 0; n < NN; ++n) bb[n] = bvec[(size_t)row * NN + n];
    float sb = bb[0] + bb[1] + bb[2] + bb[3];
    float u[4];
    float u0 = C_VAL / sb;
    #pragma unroll
    for (int i = 0; i < 4; ++i) u[i] = u0;
    float lam = 1.0f;

    for (int it = 0; it < NEWTON; ++it) {
        float e[4], Qe[4];
        #pragma unroll
        for (int i = 0; i < 4; ++i) e[i] = expf(u[i]);
        #pragma unroll
        for (int i = 0; i < 4; ++i) {
            float s = 0.0f;
            #pragma unroll
            for (int j = 0; j < 4; ++j) s += Q[i][j] * e[j];
            Qe[i] = s;
        }
        float hres = bb[0]*u[0] + bb[1]*u[1] + bb[2]*u[2] + bb[3]*u[3] - C_VAL;

        float A[5][6];                                      // augmented KKT system
        #pragma unroll
        for (int i = 0; i < 4; ++i) {
            #pragma unroll
            for (int j = 0; j < 4; ++j) A[i][j] = 2.0f * e[i] * Q[i][j] * e[j];
            A[i][i] += 2.0f * e[i] * Qe[i] + 1e-8f;
            A[i][4] = -bb[i];
            A[i][5] = -(2.0f * e[i] * Qe[i] - lam * bb[i]); // -g
        }
        #pragma unroll
        for (int j = 0; j < 4; ++j) A[4][j] = bb[j];
        A[4][4] = 0.0f;
        A[4][5] = -hres;

        // Gaussian elimination with partial pivoting (branchless row bubbling)
        #pragma unroll
        for (int c = 0; c < 5; ++c) {
            #pragma unroll
            for (int r2 = c + 1; r2 < 5; ++r2) {
                bool sw = fabsf(A[r2][c]) > fabsf(A[c][c]);
                #pragma unroll
                for (int cc = c; cc < 6; ++cc) {
                    float hi = sw ? A[r2][cc] : A[c][cc];
                    float lo = sw ? A[c][cc]  : A[r2][cc];
                    A[c][cc] = hi; A[r2][cc] = lo;
                }
            }
            float inv = 1.0f / A[c][c];
            #pragma unroll
            for (int r2 = c + 1; r2 < 5; ++r2) {
                float f = A[r2][c] * inv;
                #pragma unroll
                for (int cc = c; cc < 6; ++cc) A[r2][cc] -= f * A[c][cc];
            }
        }
        float d[5];
        #pragma unroll
        for (int c = 4; c >= 0; --c) {
            float s = A[c][5];
            #pragma unroll
            for (int cc = c + 1; cc < 5; ++cc) s -= A[c][cc] * d[cc];
            d[c] = s / A[c][c];
        }

        float nrm = sqrtf(d[0]*d[0] + d[1]*d[1] + d[2]*d[2] + d[3]*d[3] + d[4]*d[4]);
        float sc = fminf(1.0f, 10.0f / (nrm + 1e-12f));
        #pragma unroll
        for (int i = 0; i < 4; ++i) u[i] += sc * d[i];
        lam += sc * d[4];
    }

    float y[4], ys = 0.0f;
    #pragma unroll
    for (int i = 0; i < 4; ++i) { y[i] = expf(u[i]); ys += y[i]; }
    float invs = 1.0f / ys;
    #pragma unroll
    for (int i = 0; i < 4; ++i) out[(size_t)row * NN + i] = y[i] * invs;
}

// ---------------------------------------------------------------------------
// Host entry
// ---------------------------------------------------------------------------
extern "C" void kernel_launch(void* const* d_in, const int* in_sizes, int n_in,
                              void* d_out, int out_size, void* d_ws, size_t ws_size,
                              hipStream_t stream) {
    const float* x  = (const float*)d_in[0];
    const float* Qs = (const float*)d_in[1];
    const float* W1 = (const float*)d_in[2];
    const float* b1 = (const float*)d_in[3];
    const float* W2 = (const float*)d_in[4];
    const float* b2 = (const float*)d_in[5];
    float* out = (float*)d_out;

    // workspace layout: W1 bf16-hi (4MB) | W1 bf16-lo (4MB) | b vectors (256KB)
    unsigned short* w1hi = (unsigned short*)d_ws;
    unsigned short* w1lo = w1hi + (size_t)DIN * DH;
    float* bvec = (float*)(w1lo + (size_t)DIN * DH);

    const size_t smem = (size_t)2 * MTILE * PITCH * sizeof(unsigned short)
                      + (size_t)MTILE * NN * sizeof(float);   // ~132.6 KB (<< 320KB WGP LDS)
    hipFuncSetAttribute(reinterpret_cast<const void*>(fused_logits_kernel),
                        hipFuncAttributeMaxDynamicSharedMemorySize, (int)smem);

    pack_w1_kernel<<<(DIN * DH) / 256, 256, 0, stream>>>(W1, w1hi, w1lo);
    fused_logits_kernel<<<BATCH / MTILE, 256, smem, stream>>>(x, w1hi, w1lo, b1, W2, b2, bvec);
    solve_kernel<<<BATCH / 256, 256, 0, stream>>>(Qs, bvec, out);
}
